// MultiLayer_BTreeLSTM_12326556139658
// MI455X (gfx1250) — matrix operand order repaired and checked by reading, not compiled, not run on hardware
//
#include <hip/hip_runtime.h>

// ---------------------------------------------------------------------------
// MultiLayer bidirectional Tree-LSTM for MI455X (gfx1250, wave32, WMMA).
// Heap-layout tree => process level-by-level; each level is a batched GEMM
// [m,256] x [256, 6H or 5H] via v_wmma_f32_16x16x32_bf16 with the LSTM gate
// nonlinearity fused in registers.  All WMMA tile staging uses the CDNA5
// async-to-LDS path (GLOBAL_LOAD_ASYNC_TO_LDS_B128 + ASYNCcnt).
// ---------------------------------------------------------------------------

typedef __attribute__((ext_vector_type(16))) __bf16        v16bf;
typedef __attribute__((ext_vector_type(8)))  float         v8f;
typedef __attribute__((ext_vector_type(8)))  unsigned int  v8u;

#define F_DIM 512
#define H_DIM 256
#define AST   40   // LDS row stride (bf16): 80B rows => 16B-aligned b128 chunks

static __device__ __forceinline__ __bf16 f2bf(float f) {
  unsigned u = __builtin_bit_cast(unsigned, f);
  unsigned r = (u + 0x7FFFu + ((u >> 16) & 1u)) >> 16;  // round-to-nearest-even
  unsigned short s = (unsigned short)r;
  return __builtin_bit_cast(__bf16, s);
}

static __device__ __forceinline__ float sigf(float x) {
  return 1.0f / (1.0f + __expf(-x));
}

// CDNA5 async copy: 16B global -> LDS, per-lane addresses, tracked by ASYNCcnt.
static __device__ __forceinline__ void async_ld_b128(unsigned lds_off, const void* g) {
  asm volatile("global_load_async_to_lds_b128 %0, %1, off"
               :: "v"(lds_off), "v"(g) : "memory");
}
static __device__ __forceinline__ void wait_async0() {
  asm volatile("s_wait_asynccnt 0x0" ::: "memory");
}

// A fragment (16x32 MxK bf16) from LDS, row-major [16][stride].
// Contiguous-k layout used here: lane = half*16 + m; VGPR p holds a K pair:
//   K = 2*p + (e&1) + 16*half.
static __device__ __forceinline__ v16bf load_afrag(const __bf16* As, int stride) {
  int ln = threadIdx.x & 31;
  int m = ln & 15, half = ln >> 4;
  v8u t;
#pragma unroll
  for (int p = 0; p < 8; ++p) {
    int k = (p << 1) + (half << 4);
    t[p] = *(const unsigned int*)(As + m * stride + k);
  }
  return __builtin_bit_cast(v16bf, t);
}

// B fragment (32x16 KxN bf16): lanes 0-15 hold K=0..15 of col n=lane,
// lanes 16-31 hold K=16..31.  Source row-major W[col][ldk] bf16.
static __device__ __forceinline__ v16bf load_bfrag_g(const __bf16* Wb, int colBase,
                                                     int k0, int ldk) {
  int ln = threadIdx.x & 31;
  int nc = ln & 15, khalf = ln >> 4;
  const __bf16* p0 = Wb + (size_t)(colBase + nc) * ldk + k0 + (khalf << 4);
  v8u t;
#pragma unroll
  for (int p = 0; p < 8; ++p) t[p] = *(const unsigned int*)(p0 + (p << 1));
  return __builtin_bit_cast(v16bf, t);
}

static __device__ __forceinline__ v16bf load_bfrag_lds(const __bf16* Bs, int colBase,
                                                       int stride) {
  int ln = threadIdx.x & 31;
  int nc = ln & 15, khalf = ln >> 4;
  const __bf16* p0 = Bs + (colBase + nc) * stride + (khalf << 4);
  v8u t;
#pragma unroll
  for (int p = 0; p < 8; ++p) t[p] = *(const unsigned int*)(p0 + (p << 1));
  return __builtin_bit_cast(v16bf, t);
}

// ---------------------------------------------------------------------------
// C[M,NC](f32) = A[M,K](bf16) * W[NC,K](bf16)^T + bias(f32).
// 256 threads (8 waves), tile 64x128; async-to-LDS staging of both tiles.
// ---------------------------------------------------------------------------
__global__ __launch_bounds__(256) void gemm_bf16(
    const __bf16* __restrict__ A, const __bf16* __restrict__ W,
    const float* __restrict__ bias, float* __restrict__ C,
    int M, int K, int NC) {
  __shared__ __align__(16) __bf16 As[64 * AST];
  __shared__ __align__(16) __bf16 Bs[128 * AST];
  int tid = threadIdx.x;
  int w = tid >> 5, ln = tid & 31;
  int wr = w >> 1, wc = w & 1;
  int mb = blockIdx.x * 64;
  int nb = blockIdx.y * 128;

  v8f zero = {};
  v8f acc[4];
#pragma unroll
  for (int i = 0; i < 4; ++i) acc[i] = zero;

  for (int k0 = 0; k0 < K; k0 += 32) {
    __syncthreads();
    {  // A tile 64x32: 256 async b128 copies (8 bf16 each)
      int row = tid >> 2, chunk = (tid & 3) << 3;
      async_ld_b128((unsigned)(size_t)&As[row * AST + chunk],
                    A + (size_t)(mb + row) * K + k0 + chunk);
    }
    {  // B tile 128x32: 512 async b128 copies (2 per thread)
#pragma unroll
      for (int h = 0; h < 2; ++h) {
        int idx = tid + (h << 8);
        int row = idx >> 2, chunk = (idx & 3) << 3;
        async_ld_b128((unsigned)(size_t)&Bs[row * AST + chunk],
                      W + (size_t)(nb + row) * K + k0 + chunk);
      }
    }
    wait_async0();
    __syncthreads();
    v16bf a = load_afrag(As + wr * 16 * AST, AST);
#pragma unroll
    for (int nt = 0; nt < 4; ++nt) {
      v16bf b = load_bfrag_lds(Bs, wc * 64 + nt * 16, AST);
      acc[nt] = __builtin_amdgcn_wmma_f32_16x16x32_bf16(
          false, a, false, b, (short)0, acc[nt], false, false);
    }
  }
  int half = ln >> 4, nc = ln & 15;
#pragma unroll
  for (int nt = 0; nt < 4; ++nt) {
    int col = nb + wc * 64 + nt * 16 + nc;
    float bv = bias[col];
#pragma unroll
    for (int v = 0; v < 8; ++v) {
      int row = mb + wr * 16 + v + 8 * half;
      C[(size_t)row * NC + col] = acc[nt][v] + bv;
    }
  }
}

// ---------------------------------------------------------------------------
// Up scan (leaves -> root), one level at a time; block = 16 nodes.
// g = x6[node] + h[lc]@Wl^T + bl + h[rc]@Wr^T + br  (fused K=512 loop).
// h buffer is bf16 (identical rounding to converting at WMMA time) so the
// child-row gather is a pure async b128 copy into the LDS A-tile.
// Wave w owns gate columns {t*256 + w*32 .. +32} for all 6 gates.
// ---------------------------------------------------------------------------
__global__ __launch_bounds__(256) void up_level(
    const float* __restrict__ x6, const float* __restrict__ px,
    float* __restrict__ cbuf, __bf16* __restrict__ hbuf,
    const int* __restrict__ left, const int* __restrict__ right,
    const __bf16* __restrict__ WlBf, const __bf16* __restrict__ WrBf,
    const float* __restrict__ bl, const float* __restrict__ br,
    float* __restrict__ outp, int nodeBase, int cnt, int N) {
  __shared__ __align__(16) __bf16 Ah[16 * AST];
  int tid = threadIdx.x;
  int w = tid >> 5, ln = tid & 31;
  int blk16 = blockIdx.x * 16;
  int base = nodeBase + blk16;

  v8f zero = {};
  v8f acc[6][2];
#pragma unroll
  for (int t = 0; t < 6; ++t)
#pragma unroll
    for (int u = 0; u < 2; ++u) acc[t][u] = zero;

#pragma unroll 1
  for (int ks = 0; ks < 16; ++ks) {
    int k0 = ks * 32;
    __syncthreads();
    if (tid < 64) {  // gather child h rows: 64 async b128 copies
      int row = tid >> 2, chunk = (tid & 3) << 3;
      int node = base + row;
      int child = N;  // sentinel row stays zero
      if (blk16 + row < cnt) child = (k0 < 256) ? left[node] : right[node];
      async_ld_b128((unsigned)(size_t)&Ah[row * AST + chunk],
                    hbuf + (size_t)child * H_DIM + (k0 & 255) + chunk);
    }
    wait_async0();
    __syncthreads();
    v16bf a = load_afrag(Ah, AST);
    const __bf16* Wb = (k0 < 256) ? WlBf : WrBf;
    int kw = k0 & 255;
#pragma unroll
    for (int t = 0; t < 6; ++t)
#pragma unroll
      for (int u = 0; u < 2; ++u) {
        v16bf b = load_bfrag_g(Wb, t * 256 + w * 32 + u * 16, kw, 256);
        acc[t][u] = __builtin_amdgcn_wmma_f32_16x16x32_bf16(
            false, a, false, b, (short)0, acc[t][u], false, false);
      }
  }

  int half = ln >> 4, nc = ln & 15;
#pragma unroll
  for (int u = 0; u < 2; ++u) {
    int j = w * 32 + u * 16 + nc;
#pragma unroll
    for (int v = 0; v < 8; ++v) {
      int rowL = v + 8 * half;
      if (blk16 + rowL >= cnt) continue;
      int node = base + rowL;
      float g[6];
#pragma unroll
      for (int t = 0; t < 6; ++t) {
        int col = t * 256 + j;
        g[t] = acc[t][u][v] + x6[(size_t)node * 1536 + col] + bl[col] + br[col];
      }
      float gi = sigf(g[0]), go = sigf(g[1]);
      float gfl = sigf(g[2]), gfr = sigf(g[3]);
      float gu = tanhf(g[4]), gr = sigf(g[5]);
      int lc = left[node], rc = right[node];
      float c = gi * gu + gfl * cbuf[(size_t)lc * H_DIM + j]
                        + gfr * cbuf[(size_t)rc * H_DIM + j];
      float h = go * tanhf(c);
      float hf = gr * h + (1.0f - gr) * px[(size_t)node * H_DIM + j];
      cbuf[(size_t)node * H_DIM + j] = c;
      hbuf[(size_t)node * H_DIM + j] = f2bf(hf);
      outp[(size_t)node * 512 + j] = hf;  // up half: cols [0,256)
    }
  }
}

// ---------------------------------------------------------------------------
// Down scan (root -> leaves). g = x5[node] + h[parent]@Wh^T + bh (K=256).
// ---------------------------------------------------------------------------
__global__ __launch_bounds__(256) void down_level(
    const float* __restrict__ x5, const float* __restrict__ px,
    float* __restrict__ cbuf, __bf16* __restrict__ hbuf,
    const int* __restrict__ parent, const __bf16* __restrict__ WhBf,
    const float* __restrict__ bh, float* __restrict__ outp,
    int nodeBase, int cnt, int N) {
  __shared__ __align__(16) __bf16 Ah[16 * AST];
  int tid = threadIdx.x;
  int w = tid >> 5, ln = tid & 31;
  int blk16 = blockIdx.x * 16;
  int base = nodeBase + blk16;

  v8f zero = {};
  v8f acc[5][2];
#pragma unroll
  for (int t = 0; t < 5; ++t)
#pragma unroll
    for (int u = 0; u < 2; ++u) acc[t][u] = zero;

#pragma unroll 1
  for (int ks = 0; ks < 8; ++ks) {
    int k0 = ks * 32;
    __syncthreads();
    if (tid < 64) {
      int row = tid >> 2, chunk = (tid & 3) << 3;
      int node = base + row;
      int pa = (blk16 + row < cnt) ? parent[node] : N;
      async_ld_b128((unsigned)(size_t)&Ah[row * AST + chunk],
                    hbuf + (size_t)pa * H_DIM + k0 + chunk);
    }
    wait_async0();
    __syncthreads();
    v16bf a = load_afrag(Ah, AST);
#pragma unroll
    for (int t = 0; t < 5; ++t)
#pragma unroll
      for (int u = 0; u < 2; ++u) {
        v16bf b = load_bfrag_g(WhBf, t * 256 + w * 32 + u * 16, k0, 256);
        acc[t][u] = __builtin_amdgcn_wmma_f32_16x16x32_bf16(
            false, a, false, b, (short)0, acc[t][u], false, false);
      }
  }

  int half = ln >> 4, nc = ln & 15;
#pragma unroll
  for (int u = 0; u < 2; ++u) {
    int j = w * 32 + u * 16 + nc;
#pragma unroll
    for (int v = 0; v < 8; ++v) {
      int rowL = v + 8 * half;
      if (blk16 + rowL >= cnt) continue;
      int node = base + rowL;
      float g[5];
#pragma unroll
      for (int t = 0; t < 5; ++t) {
        int col = t * 256 + j;
        g[t] = acc[t][u][v] + x5[(size_t)node * 1280 + col] + bh[col];
      }
      float gi = sigf(g[0]), go = sigf(g[1]), gf = sigf(g[2]);
      float gu = tanhf(g[3]), gr = sigf(g[4]);
      int pa = parent[node];
      float c = gi * gu + gf * cbuf[(size_t)pa * H_DIM + j];
      float h = go * tanhf(c);
      float hf = gr * h + (1.0f - gr) * px[(size_t)node * H_DIM + j];
      cbuf[(size_t)node * H_DIM + j] = c;
      hbuf[(size_t)node * H_DIM + j] = f2bf(hf);
      outp[(size_t)node * 512 + 256 + j] = hf;  // down half
    }
  }
}

__global__ void cvt_f32_to_bf16(const float* __restrict__ src,
                                __bf16* __restrict__ dst, int n) {
  int i = blockIdx.x * blockDim.x + threadIdx.x;
  if (i < n) dst[i] = f2bf(src[i]);
}

// ---------------------------------------------------------------------------
extern "C" void kernel_launch(void* const* d_in, const int* in_sizes, int n_in,
                              void* d_out, int out_size, void* d_ws, size_t ws_size,
                              hipStream_t stream) {
  (void)n_in; (void)out_size; (void)ws_size;
  const float* features = (const float*)d_in[0];
  const int*   left     = (const int*)d_in[1];
  const int*   right    = (const int*)d_in[2];
  const int*   parent   = (const int*)d_in[3];
  const float* Wxf = (const float*)d_in[4];
  const float* bxf = (const float*)d_in[5];
  const float* Wlf = (const float*)d_in[6];
  const float* blf = (const float*)d_in[7];
  const float* Wrf = (const float*)d_in[8];
  const float* brf = (const float*)d_in[9];
  const float* Wpf = (const float*)d_in[10];
  const float* bpf = (const float*)d_in[11];
  const float* Wxb = (const float*)d_in[12];
  const float* bxb = (const float*)d_in[13];
  const float* Whb = (const float*)d_in[14];
  const float* bhb = (const float*)d_in[15];
  const float* Wpb = (const float*)d_in[16];
  const float* bpb = (const float*)d_in[17];

  const int N = in_sizes[0] / F_DIM;                 // 16384
  const int L = in_sizes[4] / (6 * H_DIM * F_DIM);   // 2
  const size_t fN = (size_t)N;

  // workspace carve (~195 MB)
  float*  featsA = (float*)d_ws;                     // N*512 (layer ping buffer)
  float*  gbig   = featsA + fN * 512;                // N*1536 (x6 / x5)
  float*  pxb    = gbig + fN * 1536;                 // N*256
  float*  cb     = pxb + fN * 256;                   // (N+1)*256 f32, row N = 0
  __bf16* FinBf  = (__bf16*)(cb + (fN + 1) * 256);   // N*512 bf16 layer input
  __bf16* hbBf   = FinBf + fN * 512;                 // (N+1)*256 bf16 h buffer
  __bf16* WxBf   = hbBf + (fN + 1) * 256;            // up to 1536*512
  __bf16* WpBf   = WxBf + (size_t)1536 * 512;        // 256*512
  __bf16* WlBf   = WpBf + (size_t)256 * 512;         // 1536*256
  __bf16* WrBf   = WlBf + (size_t)1536 * 256;
  __bf16* WhBf   = WrBf + (size_t)1536 * 256;        // 1280*256

  // heap-level table: level d = nodes [2^d-1, min(2^(d+1)-1, N))
  int lvBase[32], lvCnt[32], nlv = 0;
  {
    int b = 0;
    while (b < N) {
      int nxt = 2 * b + 1;
      lvBase[nlv] = b;
      lvCnt[nlv] = ((nxt < N) ? nxt : N) - b;
      ++nlv;
      b = nxt;
    }
  }

  const size_t cBytes = (fN + 1) * 256 * sizeof(float);
  const size_t hBytes = (fN + 1) * 256 * sizeof(__bf16);
  const int nF = N * 512;
  const int n1 = 1536 * 256, n2 = 1280 * 256;

#define CVT(src, dst, n) \
  cvt_f32_to_bf16<<<((n) + 255) / 256, 256, 0, stream>>>((src), (dst), (n))

  for (int l = 0; l < L; ++l) {
    const float* Fin = (l == 0) ? features : featsA;
    float* Out = (l == L - 1) ? (float*)d_out : featsA;

    CVT(Fin, FinBf, nF);
    CVT(Wlf + (size_t)l * n1, WlBf, n1);
    CVT(Wrf + (size_t)l * n1, WrBf, n1);
    CVT(Whb + (size_t)l * n2, WhBf, n2);

    // ---- up (leaves -> root) ----
    CVT(Wxf + (size_t)l * 1536 * 512, WxBf, 1536 * 512);
    CVT(Wpf + (size_t)l * 256 * 512, WpBf, 256 * 512);
    gemm_bf16<<<dim3(N / 64, 1536 / 128), 256, 0, stream>>>(
        FinBf, WxBf, bxf + (size_t)l * 1536, gbig, N, 512, 1536);
    gemm_bf16<<<dim3(N / 64, 256 / 128), 256, 0, stream>>>(
        FinBf, WpBf, bpf + (size_t)l * 256, pxb, N, 512, 256);
    hipMemsetAsync(cb, 0, cBytes, stream);
    hipMemsetAsync(hbBf, 0, hBytes, stream);
    for (int d = nlv - 1; d >= 0; --d) {
      up_level<<<(lvCnt[d] + 15) / 16, 256, 0, stream>>>(
          gbig, pxb, cb, hbBf, left, right, WlBf, WrBf,
          blf + (size_t)l * 1536, brf + (size_t)l * 1536, Out,
          lvBase[d], lvCnt[d], N);
    }

    // ---- down (root -> leaves) ----
    CVT(Wxb + (size_t)l * 1280 * 512, WxBf, 1280 * 512);
    CVT(Wpb + (size_t)l * 256 * 512, WpBf, 256 * 512);
    gemm_bf16<<<dim3(N / 64, 1280 / 128), 256, 0, stream>>>(
        FinBf, WxBf, bxb + (size_t)l * 1280, gbig, N, 512, 1280);
    gemm_bf16<<<dim3(N / 64, 256 / 128), 256, 0, stream>>>(
        FinBf, WpBf, bpb + (size_t)l * 256, pxb, N, 512, 256);
    hipMemsetAsync(cb, 0, cBytes, stream);
    hipMemsetAsync(hbBf, 0, hBytes, stream);
    for (int d = 0; d < nlv; ++d) {
      down_level<<<(lvCnt[d] + 15) / 16, 256, 0, stream>>>(
          gbig, pxb, cb, hbBf, parent, WhBf,
          bhb + (size_t)l * 1280, Out, lvBase[d], lvCnt[d], N);
    }
  }
#undef CVT
}
